// RetinaNet_21998822490131
// MI455X (gfx1250) — matrix operand order repaired and checked by reading, not compiled, not run on hardware
//
#include <hip/hip_runtime.h>
#include <hip/hip_bf16.h>
#include <math.h>

// ---------------- problem constants (match reference) ----------------
#define BB      8
#define FCH     256
#define NA      9
#define NC      80
#define IMGSZ   640
#define A_TOT   76725
#define NMS_K   1000
#define NMS_TH  0.5f

// ---------------- WMMA vector types (gfx1250, wave32) ----------------
typedef __attribute__((ext_vector_type(16))) _Float16 v16h;
typedef __attribute__((ext_vector_type(8)))  _Float16 h8;
typedef __attribute__((ext_vector_type(8)))  float    v8f;

// =====================================================================
// One-time repacks (run every launch; deterministic)
// =====================================================================

// f32 NCHW -> f16 NHWC (C = 256). i iterates in NHWC order -> coalesced store.
__global__ void nchw_to_nhwc_h(const float* __restrict__ src,
                               _Float16* __restrict__ dst, int H, int W)
{
    const int i = blockIdx.x * blockDim.x + threadIdx.x;
    const int total = BB * H * W * FCH;
    if (i >= total) return;
    const int c   = i & (FCH - 1);
    const int pix = i >> 8;                 // b*HW + y*W + x
    const int HW  = H * W;
    const int b   = pix / HW, rem = pix - b * HW;
    dst[i] = (_Float16)src[(b * FCH + c) * HW + rem];
}

// w[co][ci][ky][kx] f32 -> wp[kpos][co(padded,zero-filled)][ci] f16
__global__ void pack_w(const float* __restrict__ w, _Float16* __restrict__ wp,
                       int Cout, int CoutP)
{
    const int i = blockIdx.x * blockDim.x + threadIdx.x;
    const int total = 9 * CoutP * FCH;
    if (i >= total) return;
    const int kpos = i / (CoutP * FCH);
    const int rem  = i - kpos * (CoutP * FCH);
    const int co   = rem >> 8, ci = rem & (FCH - 1);
    const float v  = (co < Cout) ? w[(co * FCH + ci) * 9 + kpos] : 0.f;
    wp[i] = (_Float16)v;
}

// =====================================================================
// Implicit-GEMM 3x3 conv (pad=1, stride=1), Cin=256, NHWC f16 input,
// pre-packed f16 weights. Block = 256 threads = 8 waves.
// Block tile: 64 pixels x 64 out-channels. Wave tile: 16 px x 32 ch.
// Fully unrolled K loop (72 chunks) + double-buffered LDS + single
// barrier per chunk; next-chunk global loads issue before the barrier
// so their s_wait_loadcnt lands after the WMMAs of the current chunk.
// mode 0: bias+ReLU -> NHWC f16 out
// mode 1: reg final -> reg_out[(b*A_TOT + aoff + pix*NA + a)*4 + d] (f32)
// mode 2: cls final -> sigmoid -> cls_out[...*NC + c] (f32)
// =====================================================================
__global__ __launch_bounds__(256) void conv3x3_wmma(
    const _Float16* __restrict__ in,   // NHWC f16
    const _Float16* __restrict__ wp,   // [9][CoutP][256] f16
    const float* __restrict__ bias,
    _Float16* __restrict__ outH,       // mode 0
    float* __restrict__ reg_out, float* __restrict__ cls_out,
    int Cout, int CoutP, int H, int W, int mode, int aoff)
{
    const int HW   = H * W;
    const int Mtot = BB * HW;
    const int MB      = blockIdx.x * 64;
    const int co_base = blockIdx.y * 64;

    const int tid  = threadIdx.x;
    const int lane = tid & 31;
    const int wv   = tid >> 5;
    const int pw   = wv & 3;           // pixel sub-tile 0..3
    const int cw   = wv >> 2;          // channel sub-tile 0..1 (32 ch each)

    // double-buffered; stride 40 halves (80B) -> conflict-free b128 reads
    __shared__ _Float16 As[2][64 * 40];   // [pixel][ci]
    __shared__ _Float16 Bs[2][64 * 40];   // [co]   [ci]

    // ---- per-thread staging role: one pixel/co row, one 16B chunk ----
    const int srow = tid >> 2;         // 0..63
    const int scq  = (tid & 3) * 8;    // ci sub-offset (8 halves = 16B)
    const int sp   = MB + srow;
    const bool svalid = sp < Mtot;
    int sy = 0, sx = 0;
    {
        const int b = sp / HW, rem = sp - b * HW;
        sy = rem / W; sx = rem - sy * W;
    }
    const int bRowBase = (co_base + srow) * FCH;   // packed-W row for this thread

    // branchless halo load: clamp address, select-zero the result.
    // kpos/cb are compile-time constants at every call site (full unroll),
    // so ky/kx/inb/arow are CSE'd to 9 precomputed sets.
    auto loadA = [&](int kpos, int cb) -> h8 {
        const int ky = kpos / 3 - 1, kx = kpos % 3 - 1;
        const int iy = sy + ky, ix = sx + kx;
        const bool inb = svalid & (iy >= 0) & (iy < H) & (ix >= 0) & (ix < W);
        const int arow = inb ? (sp + ky * W + kx) * FCH : 0;   // always valid
        h8 av = *reinterpret_cast<const h8*>(in + arow + cb * 32 + scq);
        h8 z  = {};
        return inb ? av : z;
    };
    auto loadB = [&](int kpos, int cb) -> h8 {
        return *reinterpret_cast<const h8*>(wp + kpos * CoutP * FCH + bRowBase
                                            + cb * 32 + scq);
    };

    // warm L2 for all weight slabs of this block's channel range
    #pragma unroll
    for (int kp = 0; kp < 9; ++kp)
        __builtin_prefetch(wp + (kp * CoutP + co_base) * FCH, 0, 1);

    v8f acc0 = {}, acc1 = {};

    // fragment LDS offsets (constant per thread)
    const int mrow  = (pw << 4) + (lane & 15);
    const int k0    = (lane < 16) ? 0 : 16;
    const int kb    = (lane < 16) ? 0 : 16;
    const int nrow0 = (cw << 5) + (lane & 15);

    h8 aReg = loadA(0, 0);
    h8 bReg = loadB(0, 0);

    #pragma unroll
    for (int kpos = 0; kpos < 9; ++kpos) {
        #pragma unroll
        for (int cb = 0; cb < 8; ++cb) {
            const int k   = kpos * 8 + cb;   // compile-time constant
            const int buf = k & 1;

            *reinterpret_cast<h8*>(&As[buf][srow * 40 + scq]) = aReg;
            *reinterpret_cast<h8*>(&Bs[buf][srow * 40 + scq]) = bReg;

            // issue next-chunk fetch before the barrier (WAR on regs is
            // resolved at ds_store issue); wait lands after the WMMAs
            if (k + 1 < 72) {
                const int nk = (k + 1) / 8, ncb = (k + 1) % 8;
                aReg = loadA(nk, ncb);
                bReg = loadB(nk, ncb);
            }
            __syncthreads();

            // ---- A fragment: 16x32 f16 (contiguous k = 16*(lane/16) + s, same as B) ----
            h8 a0 = *reinterpret_cast<const h8*>(&As[buf][mrow * 40 + k0]);
            h8 a1 = *reinterpret_cast<const h8*>(&As[buf][mrow * 40 + k0 + 8]);
            v16h af = __builtin_shufflevector(a0, a1, 0,1,2,3,4,5,6,7,8,9,10,11,12,13,14,15);

            // ---- B fragments: two 32x16 columns (co +0..15, +16..31) ----
            h8 b0 = *reinterpret_cast<const h8*>(&Bs[buf][nrow0 * 40 + kb]);
            h8 b1 = *reinterpret_cast<const h8*>(&Bs[buf][nrow0 * 40 + kb + 8]);
            v16h bf0 = __builtin_shufflevector(b0, b1, 0,1,2,3,4,5,6,7,8,9,10,11,12,13,14,15);
            h8 c0 = *reinterpret_cast<const h8*>(&Bs[buf][(nrow0 + 16) * 40 + kb]);
            h8 c1 = *reinterpret_cast<const h8*>(&Bs[buf][(nrow0 + 16) * 40 + kb + 8]);
            v16h bf1 = __builtin_shufflevector(c0, c1, 0,1,2,3,4,5,6,7,8,9,10,11,12,13,14,15);

            acc0 = __builtin_amdgcn_wmma_f32_16x16x32_f16(
                       false, af, false, bf0, (short)0, acc0, false, false);
            acc1 = __builtin_amdgcn_wmma_f32_16x16x32_f16(
                       false, af, false, bf1, (short)0, acc1, false, false);
        }
    }

    // ---- epilogue: D row = v + 8*(lane>=16), col = lane&15 ----
    const int co0 = co_base + (cw << 5) + (lane & 15);
    const int co1 = co0 + 16;
    const float bv0 = (co0 < Cout) ? bias[co0] : 0.f;
    const float bv1 = (co1 < Cout) ? bias[co1] : 0.f;
    #pragma unroll
    for (int v = 0; v < 8; ++v) {
        const int p = MB + (pw << 4) + v + ((lane >> 4) << 3);
        if (p >= Mtot) continue;
        const float v0 = acc0[v] + bv0;
        const float v1 = acc1[v] + bv1;
        if (mode == 0) {
            outH[p * FCH + co0] = (_Float16)fmaxf(v0, 0.f);
            outH[p * FCH + co1] = (_Float16)fmaxf(v1, 0.f);
        } else {
            const int b = p / HW, rem = p - b * HW;
            const size_t cell = (size_t)b * A_TOT + aoff + rem * NA;
            if (mode == 1) {
                if (co0 < Cout) reg_out[(cell + (co0 >> 2)) * 4 + (co0 & 3)] = v0;
                if (co1 < Cout) reg_out[(cell + (co1 >> 2)) * 4 + (co1 & 3)] = v1;
            } else {
                if (co0 < Cout)
                    cls_out[(cell + co0 / NC) * NC + co0 % NC] = 1.f / (1.f + expf(-v0));
                if (co1 < Cout)
                    cls_out[(cell + co1 / NC) * NC + co1 % NC] = 1.f / (1.f + expf(-v1));
            }
        }
    }
}

// ============== bbox_transform + clip (elementwise) ==================
__global__ void bbox_kernel(const float* __restrict__ anchors,
                            const float* __restrict__ regs,
                            float* __restrict__ boxes)
{
    const int i = blockIdx.x * blockDim.x + threadIdx.x;
    if (i >= BB * A_TOT) return;
    const float* a = anchors + (size_t)i * 4;
    const float* d = regs    + (size_t)i * 4;
    const float w  = a[2] - a[0], h = a[3] - a[1];
    const float cx = a[0] + 0.5f * w, cy = a[1] + 0.5f * h;
    const float pcx = cx + d[0] * 0.1f * w, pcy = cy + d[1] * 0.1f * h;
    const float pw = expf(d[2] * 0.2f) * w, ph = expf(d[3] * 0.2f) * h;
    boxes[(size_t)i * 4 + 0] = fmaxf(pcx - 0.5f * pw, 0.f);
    boxes[(size_t)i * 4 + 1] = fmaxf(pcy - 0.5f * ph, 0.f);
    boxes[(size_t)i * 4 + 2] = fminf(pcx + 0.5f * pw, (float)IMGSZ);
    boxes[(size_t)i * 4 + 3] = fminf(pcy + 0.5f * ph, (float)IMGSZ);
}

// ============== per-anchor score = max over NC classes ===============
__global__ void score_kernel(const float* __restrict__ cls,
                             float* __restrict__ scores)
{
    const int i = blockIdx.x * blockDim.x + threadIdx.x;
    if (i >= BB * A_TOT) return;
    float m = -3.4e38f;
    const float* c = cls + (size_t)i * NC;
    for (int k = 0; k < NC; ++k) m = fmaxf(m, c[k]);
    scores[i] = m;
}

// ============== top-1000 per batch (iterative argmax) ================
__global__ __launch_bounds__(1024) void topk_kernel(float* __restrict__ scores,
                                                    int* __restrict__ tkidx,
                                                    float* __restrict__ idxf)
{
    __shared__ float sv[1024];
    __shared__ int   si[1024];
    const int b = blockIdx.x, tid = threadIdx.x;
    float* s = scores + (size_t)b * A_TOT;
    for (int k = 0; k < NMS_K; ++k) {
        float best = -3.4e38f; int bi = 0x7fffffff;
        for (int j = tid; j < A_TOT; j += 1024) {
            const float v = s[j];
            if (v > best || (v == best && j < bi)) { best = v; bi = j; }
        }
        sv[tid] = best; si[tid] = bi;
        __syncthreads();
        for (int st = 512; st > 0; st >>= 1) {
            if (tid < st) {
                if (sv[tid + st] > sv[tid] ||
                    (sv[tid + st] == sv[tid] && si[tid + st] < si[tid])) {
                    sv[tid] = sv[tid + st]; si[tid] = si[tid + st];
                }
            }
            __syncthreads();
        }
        if (tid == 0) {
            const int w = si[0];
            tkidx[b * NMS_K + k] = w;
            idxf [b * NMS_K + k] = (float)w;
            s[w] = -3.4e38f;
        }
        __syncthreads();
    }
}

// ============== sequential-suppression NMS per batch =================
__global__ __launch_bounds__(256) void nms_kernel(const float* __restrict__ boxes,
                                                  const int* __restrict__ tkidx,
                                                  float* __restrict__ keepf)
{
    __shared__ float X1[NMS_K], Y1[NMS_K], X2[NMS_K], Y2[NMS_K], AR[NMS_K];
    __shared__ int   act[NMS_K];
    const int b = blockIdx.x, tid = threadIdx.x;
    for (int j = tid; j < NMS_K; j += 256) {
        const int t = tkidx[b * NMS_K + j];
        const float* bx = boxes + ((size_t)b * A_TOT + t) * 4;
        const float x1 = bx[0], y1 = bx[1], x2 = bx[2], y2 = bx[3];
        X1[j] = x1; Y1[j] = y1; X2[j] = x2; Y2[j] = y2;
        AR[j] = (x2 - x1 + 1.f) * (y2 - y1 + 1.f);
        act[j] = 1;
    }
    __syncthreads();
    for (int i = 0; i < NMS_K; ++i) {
        const int ki = act[i];
        __syncthreads();
        if (tid == 0) keepf[b * NMS_K + i] = ki ? 1.f : 0.f;
        if (ki) {
            const float x1i = X1[i], y1i = Y1[i], x2i = X2[i], y2i = Y2[i], ai = AR[i];
            for (int j = tid; j < NMS_K; j += 256) {
                if (act[j]) {
                    const float xx1 = fmaxf(x1i, X1[j]), yy1 = fmaxf(y1i, Y1[j]);
                    const float xx2 = fminf(x2i, X2[j]), yy2 = fminf(y2i, Y2[j]);
                    const float iw = fmaxf(xx2 - xx1 + 1.f, 0.f);
                    const float ih = fmaxf(yy2 - yy1 + 1.f, 0.f);
                    const float inter = iw * ih;
                    if (inter / (ai + AR[j] - inter) > NMS_TH) act[j] = 0;
                }
            }
        }
        __syncthreads();
    }
}

// =====================================================================
extern "C" void kernel_launch(void* const* d_in, const int* in_sizes, int n_in,
                              void* d_out, int out_size, void* d_ws, size_t ws_size,
                              hipStream_t stream)
{
    (void)in_sizes; (void)n_in; (void)out_size; (void)ws_size;
    const float* feats[5] = { (const float*)d_in[1], (const float*)d_in[2],
                              (const float*)d_in[3], (const float*)d_in[4],
                              (const float*)d_in[5] };
    const float* anchors   = (const float*)d_in[6];
    const float* reg_w     = (const float*)d_in[7];
    const float* reg_b     = (const float*)d_in[8];
    const float* reg_out_w = (const float*)d_in[9];
    const float* reg_out_b = (const float*)d_in[10];
    const float* cls_w     = (const float*)d_in[11];
    const float* cls_b     = (const float*)d_in[12];
    const float* cls_out_w = (const float*)d_in[13];
    const float* cls_out_b = (const float*)d_in[14];

    // d_out layout: boxes | classification | keep | idx (flattened f32)
    float* boxes   = (float*)d_out;
    float* classif = boxes + (size_t)BB * A_TOT * 4;
    float* keepf   = classif + (size_t)BB * A_TOT * NC;
    float* idxf    = keepf + (size_t)BB * NMS_K;

    const int Hs[5]    = { 80, 40, 20, 10, 5 };
    const int aoffs[5] = { 0, 57600, 72000, 75600, 76500 };

    // ---------------- workspace carve (f16 first, then f32) ----------------
    _Float16* hp = (_Float16*)d_ws;
    _Float16* featH[5];
    for (int lv = 0; lv < 5; ++lv) {
        featH[lv] = hp; hp += (size_t)BB * Hs[lv] * Hs[lv] * FCH;
    }
    _Float16* bufA = hp; hp += (size_t)BB * 80 * 80 * FCH;
    _Float16* bufB = hp; hp += (size_t)BB * 80 * 80 * FCH;
    _Float16* wpReg[4], *wpCls[4];
    for (int l = 0; l < 4; ++l) { wpReg[l] = hp; hp += (size_t)9 * 256 * FCH; }
    for (int l = 0; l < 4; ++l) { wpCls[l] = hp; hp += (size_t)9 * 256 * FCH; }
    _Float16* wpRegOut = hp; hp += (size_t)9 * 64  * FCH;   // 36 -> pad 64
    _Float16* wpClsOut = hp; hp += (size_t)9 * 768 * FCH;   // 720 -> pad 768
    float* regression = (float*)hp;
    float* scores = regression + (size_t)BB * A_TOT * 4;
    int*   tkidx  = (int*)(scores + (size_t)BB * A_TOT);

    // ---------------- one-time repacks ----------------
    for (int lv = 0; lv < 5; ++lv) {
        const int n = BB * Hs[lv] * Hs[lv] * FCH;
        nchw_to_nhwc_h<<<(n + 255) / 256, 256, 0, stream>>>(feats[lv], featH[lv],
                                                            Hs[lv], Hs[lv]);
    }
    for (int l = 0; l < 4; ++l) {
        const int n = 9 * 256 * FCH;
        pack_w<<<(n + 255) / 256, 256, 0, stream>>>(reg_w + (size_t)l * FCH * FCH * 9,
                                                    wpReg[l], 256, 256);
        pack_w<<<(n + 255) / 256, 256, 0, stream>>>(cls_w + (size_t)l * FCH * FCH * 9,
                                                    wpCls[l], 256, 256);
    }
    pack_w<<<(9 * 64  * FCH + 255) / 256, 256, 0, stream>>>(reg_out_w, wpRegOut, 36, 64);
    pack_w<<<(9 * 768 * FCH + 255) / 256, 256, 0, stream>>>(cls_out_w, wpClsOut, 720, 768);

    // ---------------- conv heads ----------------
    for (int lv = 0; lv < 5; ++lv) {
        const int H = Hs[lv], W = Hs[lv];
        const int Mtot = BB * H * W;
        const int gx = (Mtot + 63) / 64;

        const _Float16* cur = featH[lv];
        for (int l = 0; l < 4; ++l) {
            _Float16* dst = (l & 1) ? bufB : bufA;
            conv3x3_wmma<<<dim3(gx, 4), 256, 0, stream>>>(
                cur, wpReg[l], reg_b + l * FCH, dst, nullptr, nullptr,
                256, 256, H, W, /*mode=*/0, 0);
            cur = dst;
        }
        conv3x3_wmma<<<dim3(gx, 1), 256, 0, stream>>>(
            cur, wpRegOut, reg_out_b, nullptr, regression, nullptr,
            36, 64, H, W, /*mode=*/1, aoffs[lv]);

        cur = featH[lv];
        for (int l = 0; l < 4; ++l) {
            _Float16* dst = (l & 1) ? bufB : bufA;
            conv3x3_wmma<<<dim3(gx, 4), 256, 0, stream>>>(
                cur, wpCls[l], cls_b + l * FCH, dst, nullptr, nullptr,
                256, 256, H, W, /*mode=*/0, 0);
            cur = dst;
        }
        conv3x3_wmma<<<dim3(gx, 12), 256, 0, stream>>>(
            cur, wpClsOut, cls_out_b, nullptr, nullptr, classif,
            720, 768, H, W, /*mode=*/2, aoffs[lv]);
    }

    const int nElem = BB * A_TOT;
    bbox_kernel <<<(nElem + 255) / 256, 256, 0, stream>>>(anchors, regression, boxes);
    score_kernel<<<(nElem + 255) / 256, 256, 0, stream>>>(classif, scores);
    topk_kernel <<<BB, 1024, 0, stream>>>(scores, tkidx, idxf);
    nms_kernel  <<<BB, 256, 0, stream>>>(boxes, tkidx, keepf);
}
